// MultiHeadAttention_46617575031101
// MI455X (gfx1250) — matrix operand order repaired and checked by reading, not compiled, not run on hardware
//
#include <hip/hip_runtime.h>
#include <hip/hip_bf16.h>

// MI455X (gfx1250) multi-head attention forward.
// Compute-bound (~69 GFLOP vs ~50MB traffic @ 23.3 TB/s) -> everything routed
// through v_wmma_f32_16x16x32_f16 (f16 inputs, f32 accumulate). Wave32 layouts
// per CDNA5 ISA 7.12.2. GEMMs use 32x64 per-wave register blocking (2x4 WMMA
// tiles) so the inner loop issues 8 WMMAs per 12 b128 loads.

typedef __attribute__((ext_vector_type(16))) _Float16 v16h;
typedef __attribute__((ext_vector_type(8)))  _Float16 v8h;
typedef __attribute__((ext_vector_type(4)))  _Float16 v4h;
typedef __attribute__((ext_vector_type(8)))  float    v8f;

static constexpr int Bn  = 2;
static constexpr int Sn  = 2048;
static constexpr int Dn  = 1024;
static constexpr int Hn  = 16;
static constexpr int DKn = 64;
static constexpr int MROWS = Bn * Sn;  // 4096

__device__ __forceinline__ int lane_id() { return threadIdx.x & 31; }

// ---------------------------------------------------------------------------
// A-operand fragment: 16x32 f16 tile from a row-major matrix, leading dim ld.
// ISA layout: lanes 0-15 -> M=lane, halves 0..7 = K 0..7, halves 8..15 = K16..23
//             lanes 16-31 -> M=lane-16, K 8..15 and K 24..31.
__device__ __forceinline__ v16h load_fragA(const _Float16* base, int ld) {
  const int lane = lane_id();
  const int row  = lane & 15;
  const int koff = (lane >> 4) << 3;  // 0 or 8
  const _Float16* p = base + (size_t)row * ld + koff;
  v8h lo = *(const v8h*)(p);
  v8h hi = *(const v8h*)(p + 16);
  v16h f;
#pragma unroll
  for (int i = 0; i < 8; ++i) { f[i] = lo[i]; f[i + 8] = hi[i]; }
  return f;
}

// B-operand fragment: 32x16 f16 (K x N). The source stores B^T row-major
// (row n = the 32 K-values of column n), leading dim ld.
// ISA layout: lanes 0-15 -> N=lane, K 0..7 and K16..23; lanes 16-31 -> N=lane-16, K 8..15 and K 24..31.
__device__ __forceinline__ v16h load_fragB(const _Float16* base, int ld) {
  const int lane = lane_id();
  const int n    = lane & 15;
  const int k0   = (lane >> 4) << 3;  // 0 or 8
  const _Float16* p = base + (size_t)n * ld + k0;
  v8h lo = *(const v8h*)(p);
  v8h hi = *(const v8h*)(p + 16);
  v16h f;
#pragma unroll
  for (int i = 0; i < 8; ++i) { f[i] = lo[i]; f[i + 8] = hi[i]; }
  return f;
}

__device__ __forceinline__ v8f wmma_f16(v16h a, v16h b, v8f c) {
  return __builtin_amdgcn_wmma_f32_16x16x32_f16(
      /*neg_a=*/false, a, /*neg_b=*/false, b,
      /*c_mod=*/(short)0, c, /*reuse_a=*/false, /*reuse_b=*/false);
}

// ---------------------------------------------------------------------------
// fp32 -> f16 conversion, 4 elements per thread (all sizes are multiples of 4)
__global__ void __launch_bounds__(256) cvt_kernel(const float* __restrict__ in,
                                                  _Float16* __restrict__ out,
                                                  int n) {
  int i = (blockIdx.x * blockDim.x + threadIdx.x) * 4;
  if (i + 3 < n) {
    float4 v = *(const float4*)(in + i);
    v4h o;
    o[0] = (_Float16)v.x; o[1] = (_Float16)v.y;
    o[2] = (_Float16)v.z; o[3] = (_Float16)v.w;
    *(v4h*)(out + i) = o;
  }
}

// ---------------------------------------------------------------------------
// Fused QKV projection: q/k/v = x @ W^T + b, stored f16 as [B, H, S, DK].
// One wave per 32x64 macro-tile (2 M-frags x 4 N-frags, 8 WMMAs / k-step).
__global__ void __launch_bounds__(256) qkv_gemm_kernel(
    const _Float16* __restrict__ xh,
    const _Float16* __restrict__ wq, const float* __restrict__ bq,
    const _Float16* __restrict__ wk, const float* __restrict__ bk,
    const _Float16* __restrict__ wv, const float* __restrict__ bv,
    _Float16* __restrict__ Qo, _Float16* __restrict__ Ko,
    _Float16* __restrict__ Vo) {
  const int wave  = threadIdx.x >> 5;
  const int tile  = blockIdx.x * 8 + wave;       // 6144 macro-tiles total
  const int mt    = tile & 127;                  // 4096/32 = 128 M tiles
  const int nt    = (tile >> 7) & 15;            // 1024/64 = 16 N tiles
  const int which = tile >> 11;                  // 0=Q 1=K 2=V
  const int m0 = mt * 32, n0 = nt * 64;

  const _Float16* W = which == 0 ? wq : (which == 1 ? wk : wv);
  const float* bias = which == 0 ? bq : (which == 1 ? bk : bv);
  _Float16* Out     = which == 0 ? Qo : (which == 1 ? Ko : Vo);

  v8f acc[2][4] = {};
#pragma unroll 2
  for (int k = 0; k < Dn; k += 32) {
    v16h a0 = load_fragA(xh + (size_t)(m0 + 0)  * Dn + k, Dn);
    v16h a1 = load_fragA(xh + (size_t)(m0 + 16) * Dn + k, Dn);
#pragma unroll
    for (int t = 0; t < 4; ++t) {
      v16h b = load_fragB(W + (size_t)(n0 + t * 16) * Dn + k, Dn);
      acc[0][t] = wmma_f16(a0, b, acc[0][t]);
      acc[1][t] = wmma_f16(a1, b, acc[1][t]);
    }
  }

  const int lane  = lane_id();
  const int rhalf = (lane >> 4) << 3;
#pragma unroll
  for (int t = 0; t < 4; ++t) {
    const int col = n0 + t * 16 + (lane & 15);
    const int h   = col >> 6;     // /DK
    const int dk  = col & 63;
    const float bb = bias[col];
#pragma unroll
    for (int mi = 0; mi < 2; ++mi) {
#pragma unroll
      for (int i = 0; i < 8; ++i) {
        const int m    = m0 + mi * 16 + i + rhalf;
        const int bidx = m >> 11;       // /S
        const int s    = m & (Sn - 1);
        Out[(((size_t)bidx * Hn + h) * Sn + s) * DKn + dk] =
            (_Float16)(acc[mi][t][i] + bb);
      }
    }
  }
}

// ---------------------------------------------------------------------------
// Causal flash attention per (b, h, 64-row query tile); 4 waves, each owns a
// 16-row strip. K tile row-major and V tile transposed are staged in LDS;
// probabilities are re-laid-out C->A through a per-wave LDS pad.
__global__ void __launch_bounds__(128) attn_kernel(
    const _Float16* __restrict__ Q, const _Float16* __restrict__ K,
    const _Float16* __restrict__ V, _Float16* __restrict__ O) {
  constexpr int LDT = 72;  // padded stride (144B = 9*16B: keeps b128 alignment)
  __shared__ _Float16 sK[64 * LDT];
  __shared__ _Float16 sVt[64 * LDT];
  __shared__ _Float16 sP[4][16 * LDT];

  const int qt   = blockIdx.x;     // 0..31 (S/64)
  const int h    = blockIdx.y;
  const int b    = blockIdx.z;
  const int wave = threadIdx.x >> 5;
  const int lane = lane_id();
  const size_t headBase = ((size_t)b * Hn + h) * (size_t)Sn * DKn;

  const int qrow0 = qt * 64 + wave * 16;

  // Q strip (16 x 64) as two A fragments, kept in registers for the whole tile
  v16h qa0 = load_fragA(Q + headBase + (size_t)qrow0 * DKn + 0, DKn);
  v16h qa1 = load_fragA(Q + headBase + (size_t)qrow0 * DKn + 32, DKn);

  float mrow[8], lrow[8];
  v8f oacc[4] = {};
#pragma unroll
  for (int i = 0; i < 8; ++i) { mrow[i] = -3.0e38f; lrow[i] = 0.0f; }

  const int colbase = lane & 15;
  const int rhalf   = (lane >> 4) << 3;

  for (int j = 0; j <= qt; ++j) {  // causal: key blocks up to the diagonal
    const int kb = j * 64;

    // ---- cooperative staging: sK row-major, sVt transposed ----
    {
      const int row = threadIdx.x >> 1;        // 0..63
      const int c0  = (threadIdx.x & 1) * 32;  // 0 or 32
      const _Float16* kg = K + headBase + (size_t)(kb + row) * DKn + c0;
      const _Float16* vg = V + headBase + (size_t)(kb + row) * DKn + c0;
      v8h k0 = *(const v8h*)(kg + 0);
      v8h k1 = *(const v8h*)(kg + 8);
      v8h k2 = *(const v8h*)(kg + 16);
      v8h k3 = *(const v8h*)(kg + 24);
      *(v8h*)(&sK[row * LDT + c0 + 0])  = k0;
      *(v8h*)(&sK[row * LDT + c0 + 8])  = k1;
      *(v8h*)(&sK[row * LDT + c0 + 16]) = k2;
      *(v8h*)(&sK[row * LDT + c0 + 24]) = k3;
      v8h v0 = *(const v8h*)(vg + 0);
      v8h v1 = *(const v8h*)(vg + 8);
      v8h v2 = *(const v8h*)(vg + 16);
      v8h v3 = *(const v8h*)(vg + 24);
#pragma unroll
      for (int e = 0; e < 8; ++e) {
        sVt[(c0 + 0  + e) * LDT + row] = v0[e];
        sVt[(c0 + 8  + e) * LDT + row] = v1[e];
        sVt[(c0 + 16 + e) * LDT + row] = v2[e];
        sVt[(c0 + 24 + e) * LDT + row] = v3[e];
      }
    }
    __syncthreads();

    // ---- scores S = Q K^T : 16x64 per wave (8 WMMAs) ----
    v8f sacc[4] = {};
#pragma unroll
    for (int t = 0; t < 4; ++t) {
      v16h b0 = load_fragB(&sK[(t * 16) * LDT + 0], LDT);
      sacc[t] = wmma_f16(qa0, b0, sacc[t]);
      v16h b1 = load_fragB(&sK[(t * 16) * LDT + 32], LDT);
      sacc[t] = wmma_f16(qa1, b1, sacc[t]);
    }

    // ---- online softmax (f32). Row r lives in v8f slot r%8 of one 16-lane
    //      half; shfl_xor with masks <16 stays inside each 16-lane group. ----
#pragma unroll
    for (int i = 0; i < 8; ++i) {
      const int qrow = qrow0 + i + rhalf;
      float mx = mrow[i];
#pragma unroll
      for (int t = 0; t < 4; ++t) {
        const int kcol = kb + t * 16 + colbase;
        float sc = sacc[t][i] * 0.125f;  // 1/sqrt(64)
        sc = (kcol <= qrow) ? sc : -3.0e38f;
        sacc[t][i] = sc;
        mx = fmaxf(mx, sc);
      }
#pragma unroll
      for (int off = 1; off < 16; off <<= 1)
        mx = fmaxf(mx, __shfl_xor(mx, off, 32));
      const float resc = __expf(mrow[i] - mx);
      float psum = 0.0f;
#pragma unroll
      for (int t = 0; t < 4; ++t) {
        float pv = __expf(sacc[t][i] - mx);
        sacc[t][i] = pv;
        psum += pv;
      }
#pragma unroll
      for (int off = 1; off < 16; off <<= 1)
        psum += __shfl_xor(psum, off, 32);
      lrow[i] = lrow[i] * resc + psum;
      mrow[i] = mx;
#pragma unroll
      for (int t = 0; t < 4; ++t) oacc[t][i] *= resc;
    }

    // ---- P: C-layout -> A-fragments via per-wave LDS pad (same-wave DS ops
    //      are in-order, no barrier needed) ----
#pragma unroll
    for (int i = 0; i < 8; ++i)
#pragma unroll
      for (int t = 0; t < 4; ++t)
        sP[wave][(i + rhalf) * LDT + t * 16 + colbase] = (_Float16)sacc[t][i];

    v16h pa0 = load_fragA(&sP[wave][0], LDT);
    v16h pa1 = load_fragA(&sP[wave][32], LDT);

    // ---- O += P V : B fragments read V^T rows from sVt (8 WMMAs) ----
#pragma unroll
    for (int t = 0; t < 4; ++t) {
      v16h b0 = load_fragB(&sVt[(t * 16) * LDT + 0], LDT);
      oacc[t] = wmma_f16(pa0, b0, oacc[t]);
      v16h b1 = load_fragB(&sVt[(t * 16) * LDT + 32], LDT);
      oacc[t] = wmma_f16(pa1, b1, oacc[t]);
    }
    __syncthreads();  // protect sK/sVt before next stage
  }

  // ---- normalize, store concat-heads layout [B, S, D] (f16) ----
#pragma unroll
  for (int i = 0; i < 8; ++i) {
    const float inv = 1.0f / lrow[i];
    const int srow  = qrow0 + i + rhalf;
#pragma unroll
    for (int t = 0; t < 4; ++t) {
      O[((size_t)b * Sn + srow) * Dn + h * DKn + t * 16 + colbase] =
          (_Float16)(oacc[t][i] * inv);
    }
  }
}

// ---------------------------------------------------------------------------
// Output projection: out = attnO @ Wo^T + bo, fp32 result.
// One wave per 32x64 macro-tile (2x4 WMMA register blocking).
__global__ void __launch_bounds__(256) out_gemm_kernel(
    const _Float16* __restrict__ A, const _Float16* __restrict__ wo,
    const float* __restrict__ bo, float* __restrict__ out) {
  const int wave = threadIdx.x >> 5;
  const int tile = blockIdx.x * 8 + wave;   // 2048 macro-tiles
  const int mt = tile & 127;                // 4096/32
  const int nt = tile >> 7;                 // 1024/64
  const int m0 = mt * 32, n0 = nt * 64;

  v8f acc[2][4] = {};
#pragma unroll 2
  for (int k = 0; k < Dn; k += 32) {
    v16h a0 = load_fragA(A + (size_t)(m0 + 0)  * Dn + k, Dn);
    v16h a1 = load_fragA(A + (size_t)(m0 + 16) * Dn + k, Dn);
#pragma unroll
    for (int t = 0; t < 4; ++t) {
      v16h b = load_fragB(wo + (size_t)(n0 + t * 16) * Dn + k, Dn);
      acc[0][t] = wmma_f16(a0, b, acc[0][t]);
      acc[1][t] = wmma_f16(a1, b, acc[1][t]);
    }
  }

  const int lane  = lane_id();
  const int rhalf = (lane >> 4) << 3;
#pragma unroll
  for (int t = 0; t < 4; ++t) {
    const int col  = n0 + t * 16 + (lane & 15);
    const float bb = bo[col];
#pragma unroll
    for (int mi = 0; mi < 2; ++mi) {
#pragma unroll
      for (int i = 0; i < 8; ++i) {
        const int m = m0 + mi * 16 + i + rhalf;
        out[(size_t)m * Dn + col] = acc[mi][t][i] + bb;
      }
    }
  }
}

// ---------------------------------------------------------------------------
extern "C" void kernel_launch(void* const* d_in, const int* in_sizes, int n_in,
                              void* d_out, int out_size, void* d_ws,
                              size_t ws_size, hipStream_t stream) {
  (void)in_sizes; (void)n_in; (void)out_size; (void)ws_size;
  const float* x  = (const float*)d_in[0];
  const float* Wq = (const float*)d_in[1];
  const float* bq = (const float*)d_in[2];
  const float* Wk = (const float*)d_in[3];
  const float* bk = (const float*)d_in[4];
  const float* Wv = (const float*)d_in[5];
  const float* bv = (const float*)d_in[6];
  const float* Wo = (const float*)d_in[7];
  const float* bo = (const float*)d_in[8];
  float* out = (float*)d_out;

  // Workspace carve (f16): x, 4 weights, Q, K, V, attn-out  => ~48 MB
  _Float16* p   = (_Float16*)d_ws;
  _Float16* xh  = p; p += (size_t)MROWS * Dn;
  _Float16* wqh = p; p += (size_t)Dn * Dn;
  _Float16* wkh = p; p += (size_t)Dn * Dn;
  _Float16* wvh = p; p += (size_t)Dn * Dn;
  _Float16* woh = p; p += (size_t)Dn * Dn;
  _Float16* Qb  = p; p += (size_t)MROWS * Dn;
  _Float16* Kb  = p; p += (size_t)MROWS * Dn;
  _Float16* Vb  = p; p += (size_t)MROWS * Dn;
  _Float16* Ob  = p; p += (size_t)MROWS * Dn;

  // 1) fp32 -> f16 conversions
  {
    const int nx = MROWS * Dn;
    cvt_kernel<<<(nx / 4 + 255) / 256, 256, 0, stream>>>(x, xh, nx);
    const int nw = Dn * Dn;
    cvt_kernel<<<(nw / 4 + 255) / 256, 256, 0, stream>>>(Wq, wqh, nw);
    cvt_kernel<<<(nw / 4 + 255) / 256, 256, 0, stream>>>(Wk, wkh, nw);
    cvt_kernel<<<(nw / 4 + 255) / 256, 256, 0, stream>>>(Wv, wvh, nw);
    cvt_kernel<<<(nw / 4 + 255) / 256, 256, 0, stream>>>(Wo, woh, nw);
  }

  // 2) fused QKV projection: 3 * 128 * 16 = 6144 wave-tiles / 8 per block
  qkv_gemm_kernel<<<768, 256, 0, stream>>>(xh, wqh, bq, wkh, bk, wvh, bv,
                                           Qb, Kb, Vb);

  // 3) causal flash attention
  attn_kernel<<<dim3(Sn / 64, Hn, Bn), 128, 0, stream>>>(Qb, Kb, Vb, Ob);

  // 4) output projection: 128 * 16 = 2048 wave-tiles / 8 per block
  out_gemm_kernel<<<256, 256, 0, stream>>>(Ob, woh, bo, out);
}
